// Policy_1546188227218
// MI455X (gfx1250) — matrix operand order repaired and checked by reading, not compiled, not run on hardware
//
#include <hip/hip_runtime.h>
#include <math.h>

#define HID   128
#define NHEAD 8
#define HDIM  16
#define LHD   128
#define NA    7
#define NO    2

typedef __attribute__((ext_vector_type(16))) _Float16 v16h;
typedef __attribute__((ext_vector_type(8)))  float    v8f;
typedef __attribute__((ext_vector_type(4)))  float    f32x4;

__device__ __forceinline__ float sigmoidf_(float x) { return 1.0f / (1.0f + expf(-x)); }

// float atomic max via sign-split integer atomics (works for any mix of signs)
__device__ __forceinline__ void atomicMaxFloat(float* addr, float val) {
    if (val >= 0.0f) atomicMax((int*)addr, __float_as_int(val));
    else             atomicMin((unsigned int*)addr, __float_as_uint(val));
}

// ---------------- generic fill ----------------
__global__ void k_fill(float* __restrict__ p, float v, int n) {
    int i = blockIdx.x * blockDim.x + threadIdx.x;
    if (i < n) p[i] = v;
}

// ---------------- degree / norm ----------------
__global__ void k_deg(const int* __restrict__ dst, float* __restrict__ deg, int E) {
    int e = blockIdx.x * blockDim.x + threadIdx.x;
    if (e < E) atomicAdd(&deg[dst[e]], 1.0f);
}
__global__ void k_dinv(float* __restrict__ deg, int N) {
    int i = blockIdx.x * blockDim.x + threadIdx.x;
    if (i < N) deg[i] = rsqrtf(fmaxf(deg[i], 1.0f));   // deg >= 1 (self-loop)
}

// ---------------- WMMA GEMM: Y[N x 128] = act(X[N x K] @ W[128 x K]^T + b) ----------------
// One wave per 16x16 output tile; K templated (32 or 128) so the K-loop fully
// unrolls into back-to-back v_wmma ops with b128 fragment loads.
template <int K>
__global__ __launch_bounds__(128)
void k_gemm(const float* __restrict__ X, const float* __restrict__ W,
            const float* __restrict__ bias, float* __restrict__ Y,
            int N, int relu_f) {
    const int tiles_m = N >> 4;
    const int tile = blockIdx.x * 4 + (threadIdx.x >> 5);
    if (tile >= tiles_m * (HID / 16)) return;       // whole wave exits together
    const int tm = tile % tiles_m;
    const int tn = tile / tiles_m;
    const int lane = threadIdx.x & 31;
    const int hi = lane >> 4;                       // half-wave select
    const int l15 = lane & 15;

    const float* xrow = X + (size_t)(tm * 16 + l15) * K;  // A: M = l15 (both halves)
    const float* wrow = W + (size_t)(tn * 16 + l15) * K;  // B: N = l15 (both halves)

    v8f acc = {};
#pragma unroll
    for (int kb = 0; kb < K; kb += 32) {
        // 16-bit A 16x32 layout: lane half 'hi' needs K runs [kb+8hi, +8) and [kb+16+8hi, +8)
        // 16-bit B 32x16 layout: lane half 'hi' needs K runs [kb+8hi, +8) and [kb+16+8hi, +8)
        const f32x4* ap0 = (const f32x4*)(xrow + kb + hi * 8);
        const f32x4* ap1 = (const f32x4*)(xrow + kb + 16 + hi * 8);
        const f32x4* bp0 = (const f32x4*)(wrow + kb + hi * 8);
        const f32x4* bp1 = (const f32x4*)(wrow + kb + 16 + hi * 8);
        f32x4 a0 = ap0[0], a1 = ap0[1];
        f32x4 a2 = ap1[0], a3 = ap1[1];
        f32x4 b0 = bp0[0], b1 = bp0[1], b2 = bp1[0], b3 = bp1[1];
        v16h a, b;
#pragma unroll
        for (int j = 0; j < 4; ++j) {
            a[j]      = (_Float16)a0[j];
            a[4 + j]  = (_Float16)a1[j];
            a[8 + j]  = (_Float16)a2[j];
            a[12 + j] = (_Float16)a3[j];
            b[j]      = (_Float16)b0[j];
            b[4 + j]  = (_Float16)b1[j];
            b[8 + j]  = (_Float16)b2[j];
            b[12 + j] = (_Float16)b3[j];
        }
        acc = __builtin_amdgcn_wmma_f32_16x16x32_f16(
            /*neg_a=*/false, a, /*neg_b=*/false, b,
            /*c_mod=*/(short)0, acc, /*reuse_a=*/false, /*reuse_b=*/false);
    }
    const int col = tn * 16 + l15;
    const float bb = bias ? bias[col] : 0.0f;
#pragma unroll
    for (int r = 0; r < 8; ++r) {                   // D: VGPR r -> M = r + hi*8
        float o = acc[r] + bb;
        if (relu_f) o = fmaxf(o, 0.0f);
        Y[(size_t)(tm * 16 + r + hi * 8) * HID + col] = o;
    }
}

// ---------------- GCN edge aggregation: agg[dst] += dinv[s]*dinv[d]*xw[src] ----------------
__global__ void k_gcn_edges(const int* __restrict__ src, const int* __restrict__ dst,
                            const float* __restrict__ dinv, const float* __restrict__ xw,
                            float* __restrict__ agg, int E) {
    int idx = blockIdx.x * blockDim.x + threadIdx.x;
    int e = idx >> 7, c = idx & 127;
    if (e >= E) return;
    int s = src[e], d = dst[e];
    float nrm = dinv[s] * dinv[d];
    atomicAdd(&agg[(size_t)d * HID + c], nrm * xw[(size_t)s * HID + c]);
}

// out = relu(agg + dinv^2 * xw (self loop) + bias)
__global__ void k_gcn_finish(const float* __restrict__ agg, const float* __restrict__ xw,
                             const float* __restrict__ dinv, const float* __restrict__ bias,
                             float* __restrict__ out, int N) {
    int idx = blockIdx.x * blockDim.x + threadIdx.x;
    int i = idx >> 7, c = idx & 127;
    if (i >= N) return;
    float di = dinv[i];
    out[idx] = fmaxf(agg[idx] + di * di * xw[idx] + bias[c], 0.0f);
}

// ---------------- attention ----------------
__global__ void k_attn_logits(const int* __restrict__ src, const int* __restrict__ dst,
                              const float* __restrict__ q, const float* __restrict__ k,
                              float* __restrict__ L, float* __restrict__ m, int E) {
    int idx = blockIdx.x * blockDim.x + threadIdx.x;
    int e = idx >> 3, h = idx & 7;
    if (e >= E) return;
    int d = dst[e], s = src[e];
    const f32x4* qd = (const f32x4*)(q + (size_t)d * HID + h * HDIM);
    const f32x4* ks = (const f32x4*)(k + (size_t)s * HID + h * HDIM);
    float acc = 0.0f;
#pragma unroll
    for (int j = 0; j < 4; ++j) {
        f32x4 qq = qd[j], kk = ks[j];
        acc += qq[0] * kk[0] + qq[1] * kk[1] + qq[2] * kk[2] + qq[3] * kk[3];
    }
    acc *= 0.25f;                                    // 1/sqrt(16)
    L[(size_t)e * NHEAD + h] = acc;
    atomicMaxFloat(&m[(size_t)d * NHEAD + h], acc);
}

__global__ void k_m_fix(float* __restrict__ m, int n) {
    int i = blockIdx.x * blockDim.x + threadIdx.x;
    if (i < n) { float v = m[i]; if (!(v > -3.0e38f && v < 3.0e38f)) m[i] = 0.0f; }
}

__global__ void k_attn_exp(const int* __restrict__ dst, float* __restrict__ L,
                           const float* __restrict__ m, float* __restrict__ den, int E) {
    int idx = blockIdx.x * blockDim.x + threadIdx.x;
    int e = idx >> 3, h = idx & 7;
    if (e >= E) return;
    int d = dst[e];
    float ex = expf(L[(size_t)e * NHEAD + h] - m[(size_t)d * NHEAD + h]);
    L[(size_t)e * NHEAD + h] = ex;
    atomicAdd(&den[(size_t)d * NHEAD + h], ex);
}

__global__ void k_attn_agg(const int* __restrict__ src, const int* __restrict__ dst,
                           const float* __restrict__ L, const float* __restrict__ den,
                           const float* __restrict__ v, float* __restrict__ agg, int E) {
    int idx = blockIdx.x * blockDim.x + threadIdx.x;
    int e = idx >> 7, c = idx & 127;
    if (e >= E) return;
    int h = c >> 4;
    int d = dst[e], s = src[e];
    float alpha = L[(size_t)e * NHEAD + h] / fmaxf(den[(size_t)d * NHEAD + h], 1e-16f);
    atomicAdd(&agg[(size_t)d * HID + c], alpha * v[(size_t)s * HID + c]);
}

// ---------------- FiLM params (B x 128) ----------------
__global__ void k_film(const float* __restrict__ w,
                       const float* __restrict__ Wg, const float* __restrict__ bg,
                       const float* __restrict__ Wb, const float* __restrict__ bb,
                       float* __restrict__ gam, float* __restrict__ bet, int B) {
    int idx = blockIdx.x * blockDim.x + threadIdx.x;
    int b = idx >> 7, c = idx & 127;
    if (b >= B) return;
    float w0 = w[b * NO + 0], w1 = w[b * NO + 1];
    gam[idx] = bg[c] + w0 * Wg[c * NO + 0] + w1 * Wg[c * NO + 1];
    bet[idx] = bb[c] + w0 * Wb[c * NO + 0] + w1 * Wb[c * NO + 1];
}

__global__ void k_cnt(const int* __restrict__ batch, float* __restrict__ cnt, int N) {
    int i = blockIdx.x * blockDim.x + threadIdx.x;
    if (i < N) atomicAdd(&cnt[batch[i]], 1.0f);
}

// x = (agg + skip)*(1+gam[b]) + bet[b]; pooled sum per graph (FiLM never materialized)
__global__ void k_film_pool(const float* __restrict__ agg, const float* __restrict__ skip,
                            const int* __restrict__ batch,
                            const float* __restrict__ gam, const float* __restrict__ bet,
                            float* __restrict__ S, int N) {
    int idx = blockIdx.x * blockDim.x + threadIdx.x;
    int i = idx >> 7, c = idx & 127;
    if (i >= N) return;
    int b = batch[i];
    float xf = agg[idx] + skip[idx];
    xf = xf * (1.0f + gam[b * HID + c]) + bet[b * HID + c];
    atomicAdd(&S[(size_t)b * HID + c], xf);
}

// ---------------- LSTM gates: [B x 512] ----------------
__global__ void k_gates(const float* __restrict__ S, const float* __restrict__ cnt,
                        const float* __restrict__ hprev,
                        const float* __restrict__ Wih, const float* __restrict__ bih,
                        const float* __restrict__ Whh, const float* __restrict__ bhh,
                        float* __restrict__ gates, int B) {
    int idx = blockIdx.x * blockDim.x + threadIdx.x;
    int r = idx >> 9, g = idx & 511;
    if (r >= B) return;
    float invc = 1.0f / fmaxf(cnt[r], 1.0f);
    float acc = bih[g] + bhh[g];
    const float* sr = S + (size_t)r * HID;
    const float* wi = Wih + (size_t)g * HID;
    for (int c = 0; c < HID; ++c) acc += (sr[c] * invc) * wi[c];
    const float* hr = hprev + (size_t)r * LHD;
    const float* wh = Whh + (size_t)g * LHD;
    for (int c = 0; c < LHD; ++c) acc += hr[c] * wh[c];
    gates[idx] = acc;
}

// ---------------- LSTM elementwise + LayerNorm; writes h_new (ws+out), c_new (out) ----------------
__global__ __launch_bounds__(128)
void k_lstm_ln(const float* __restrict__ gates, const float* __restrict__ cprev,
               const float* __restrict__ lng, const float* __restrict__ lnb,
               float* __restrict__ hnew_ws, float* __restrict__ out_h, float* __restrict__ out_c) {
    __shared__ float red[LHD];
    __shared__ float mu_s, var_s;
    int r = blockIdx.x, c = threadIdx.x;
    const float* gr = gates + (size_t)r * 4 * LHD;
    float ig = sigmoidf_(gr[c]);
    float fg = sigmoidf_(gr[LHD + c]);
    float gg = tanhf(gr[2 * LHD + c]);
    float og = sigmoidf_(gr[3 * LHD + c]);
    float craw = fg * cprev[(size_t)r * LHD + c] + ig * gg;
    float hraw = og * tanhf(craw);                       // h uses unclamped c (as in source)
    float cnew = fminf(fmaxf(craw, -1e6f), 1e6f);

    red[c] = hraw; __syncthreads();
    for (int s = LHD / 2; s > 0; s >>= 1) { if (c < s) red[c] += red[c + s]; __syncthreads(); }
    if (c == 0) mu_s = red[0] * (1.0f / LHD);
    __syncthreads();
    float dvar = hraw - mu_s;
    red[c] = dvar * dvar; __syncthreads();
    for (int s = LHD / 2; s > 0; s >>= 1) { if (c < s) red[c] += red[c + s]; __syncthreads(); }
    if (c == 0) var_s = red[0] * (1.0f / LHD);
    __syncthreads();
    float hn = (hraw - mu_s) * rsqrtf(var_s + 1e-5f) * lng[c] + lnb[c];
    hnew_ws[(size_t)r * LHD + c] = hn;
    out_h[(size_t)r * LHD + c] = hn;
    out_c[(size_t)r * LHD + c] = cnew;
}

// out[r][c] = relu(b[c] + dot(h[r], W[c]))  (rows=B, 128->128)
__global__ void k_head_mm(const float* __restrict__ h, const float* __restrict__ W,
                          const float* __restrict__ b, float* __restrict__ out, int B) {
    int idx = blockIdx.x * blockDim.x + threadIdx.x;
    int r = idx >> 7, c = idx & 127;
    if (r >= B) return;
    float acc = b[c];
    const float* hr = h + (size_t)r * LHD;
    const float* wr = W + (size_t)c * LHD;
    for (int j = 0; j < LHD; ++j) acc += hr[j] * wr[j];
    out[idx] = fmaxf(acc, 0.0f);
}

// dueling head combine -> qv [B x 7 x 2]
__global__ void k_final(const float* __restrict__ hv1, const float* __restrict__ ha1,
                        const float* __restrict__ Wv2, const float* __restrict__ bv2,
                        const float* __restrict__ Wa2, const float* __restrict__ ba2,
                        float* __restrict__ out_q, int B) {
    int r = blockIdx.x * blockDim.x + threadIdx.x;
    if (r >= B) return;
    const float* hv = hv1 + (size_t)r * LHD;
    const float* ha = ha1 + (size_t)r * LHD;
    float val[NO];
    for (int o = 0; o < NO; ++o) {
        float acc = bv2[o];
        const float* wr = Wv2 + (size_t)o * LHD;
        for (int j = 0; j < LHD; ++j) acc += hv[j] * wr[j];
        val[o] = acc;
    }
    float adv[NA * NO], mean[NO] = {0.0f, 0.0f};
    for (int a = 0; a < NA; ++a)
        for (int o = 0; o < NO; ++o) {
            float acc = ba2[a * NO + o];
            const float* wr = Wa2 + (size_t)(a * NO + o) * LHD;
            for (int j = 0; j < LHD; ++j) acc += ha[j] * wr[j];
            adv[a * NO + o] = acc;
            mean[o] += acc;
        }
    mean[0] *= (1.0f / NA); mean[1] *= (1.0f / NA);
    for (int a = 0; a < NA; ++a)
        for (int o = 0; o < NO; ++o) {
            float qq = val[o] + adv[a * NO + o] - mean[o];
            if (qq != qq) qq = 0.0f;                     // nan_to_num
            qq = fminf(fmaxf(qq, -100.0f), 100.0f);      // clip (covers +-inf)
            out_q[(size_t)r * NA * NO + a * NO + o] = qq;
        }
}

// =====================================================================
extern "C" void kernel_launch(void* const* d_in, const int* in_sizes, int n_in,
                              void* d_out, int out_size, void* d_ws, size_t ws_size,
                              hipStream_t stream) {
    (void)n_in; (void)out_size; (void)ws_size;
    const float* nf     = (const float*)d_in[0];
    const int*   ei     = (const int*)  d_in[1];
    const int*   batch  = (const int*)  d_in[2];
    const float* wobj   = (const float*)d_in[3];
    const float* hprev  = (const float*)d_in[4];
    const float* cprev  = (const float*)d_in[5];
    const float* W_emb  = (const float*)d_in[6];  const float* b_emb = (const float*)d_in[7];
    const float* W_g1   = (const float*)d_in[8];  const float* b_g1  = (const float*)d_in[9];
    const float* W_g2   = (const float*)d_in[10]; const float* b_g2  = (const float*)d_in[11];
    const float* Wq     = (const float*)d_in[12]; const float* bq    = (const float*)d_in[13];
    const float* Wk     = (const float*)d_in[14]; const float* bk    = (const float*)d_in[15];
    const float* Wv     = (const float*)d_in[16]; const float* bv    = (const float*)d_in[17];
    const float* Wskip  = (const float*)d_in[18]; const float* bskip = (const float*)d_in[19];
    const float* W_gam  = (const float*)d_in[20]; const float* b_gam = (const float*)d_in[21];
    const float* W_bet  = (const float*)d_in[22]; const float* b_bet = (const float*)d_in[23];
    const float* W_ih   = (const float*)d_in[24]; const float* b_ih  = (const float*)d_in[25];
    const float* W_hh   = (const float*)d_in[26]; const float* b_hh  = (const float*)d_in[27];
    const float* ln_g   = (const float*)d_in[28]; const float* ln_b  = (const float*)d_in[29];
    const float* W_a1   = (const float*)d_in[30]; const float* b_a1  = (const float*)d_in[31];
    const float* W_a2   = (const float*)d_in[32]; const float* b_a2  = (const float*)d_in[33];
    const float* W_v1   = (const float*)d_in[34]; const float* b_v1  = (const float*)d_in[35];
    const float* W_v2   = (const float*)d_in[36]; const float* b_v2  = (const float*)d_in[37];

    const int N = in_sizes[0] / 32;      // 50000
    const int E = in_sizes[1] / 2;       // 1600000
    const int B = in_sizes[4] / LHD;     // 64
    const int* src = ei;
    const int* dst = ei + E;

    // workspace carve-up (floats)
    float* ws = (float*)d_ws;
    size_t o = 0;
    float* X   = ws + o; o += (size_t)N * HID;
    float* T   = ws + o; o += (size_t)N * HID;   // xw / skip
    float* AG  = ws + o; o += (size_t)N * HID;
    float* Q   = ws + o; o += (size_t)N * HID;
    float* Kb  = ws + o; o += (size_t)N * HID;
    float* V   = ws + o; o += (size_t)N * HID;
    float* DEG = ws + o; o += (size_t)N;         // becomes dinv
    float* M   = ws + o; o += (size_t)N * NHEAD;
    float* DEN = ws + o; o += (size_t)N * NHEAD;
    float* L   = ws + o; o += (size_t)E * NHEAD; // logits -> exp
    float* GAM = ws + o; o += (size_t)B * HID;
    float* BET = ws + o; o += (size_t)B * HID;
    float* S   = ws + o; o += (size_t)B * HID;
    float* CNT = ws + o; o += (size_t)B;
    float* GT  = ws + o; o += (size_t)B * 4 * LHD;
    float* HN  = ws + o; o += (size_t)B * LHD;
    float* HV1 = ws + o; o += (size_t)B * LHD;
    float* HA1 = ws + o; o += (size_t)B * LHD;

    float* out_q = (float*)d_out;                       // 896
    float* out_h = out_q + (size_t)B * NA * NO;         // 8192
    float* out_c = out_h + (size_t)B * LHD;             // 8192

    auto cdiv = [](long long a, long long b) { return (unsigned)((a + b - 1) / b); };
    const int tiles = (N / 16) * (HID / 16);
    const unsigned ggemm = cdiv(tiles, 4);
    const float NEG_INF = -__builtin_inff();

    // ---- GCN normalization ----
    k_fill<<<cdiv(N, 256), 256, 0, stream>>>(DEG, 1.0f, N);          // self-loop
    k_deg<<<cdiv(E, 256), 256, 0, stream>>>(dst, DEG, E);
    k_dinv<<<cdiv(N, 256), 256, 0, stream>>>(DEG, N);

    // ---- embed ----
    k_gemm<32><<<ggemm, 128, 0, stream>>>(nf, W_emb, b_emb, X, N, 1);

    // ---- GCN layer 1 ----
    k_gemm<128><<<ggemm, 128, 0, stream>>>(X, W_g1, nullptr, T, N, 0);
    k_fill<<<cdiv((size_t)N * HID, 256), 256, 0, stream>>>(AG, 0.0f, N * HID);
    k_gcn_edges<<<cdiv((size_t)E * HID, 256), 256, 0, stream>>>(src, dst, DEG, T, AG, E);
    k_gcn_finish<<<cdiv((size_t)N * HID, 256), 256, 0, stream>>>(AG, T, DEG, b_g1, X, N);

    // ---- GCN layer 2 ----
    k_gemm<128><<<ggemm, 128, 0, stream>>>(X, W_g2, nullptr, T, N, 0);
    k_fill<<<cdiv((size_t)N * HID, 256), 256, 0, stream>>>(AG, 0.0f, N * HID);
    k_gcn_edges<<<cdiv((size_t)E * HID, 256), 256, 0, stream>>>(src, dst, DEG, T, AG, E);
    k_gcn_finish<<<cdiv((size_t)N * HID, 256), 256, 0, stream>>>(AG, T, DEG, b_g2, X, N);

    // ---- TransformerConv projections ----
    k_gemm<128><<<ggemm, 128, 0, stream>>>(X, Wq, bq, Q, N, 0);
    k_gemm<128><<<ggemm, 128, 0, stream>>>(X, Wk, bk, Kb, N, 0);
    k_gemm<128><<<ggemm, 128, 0, stream>>>(X, Wv, bv, V, N, 0);
    k_gemm<128><<<ggemm, 128, 0, stream>>>(X, Wskip, bskip, T, N, 0);  // skip path

    // ---- edge softmax + aggregate ----
    k_fill<<<cdiv((size_t)N * NHEAD, 256), 256, 0, stream>>>(M, NEG_INF, N * NHEAD);
    k_fill<<<cdiv((size_t)N * NHEAD, 256), 256, 0, stream>>>(DEN, 0.0f, N * NHEAD);
    k_attn_logits<<<cdiv((size_t)E * NHEAD, 256), 256, 0, stream>>>(src, dst, Q, Kb, L, M, E);
    k_m_fix<<<cdiv((size_t)N * NHEAD, 256), 256, 0, stream>>>(M, N * NHEAD);
    k_attn_exp<<<cdiv((size_t)E * NHEAD, 256), 256, 0, stream>>>(dst, L, M, DEN, E);
    k_fill<<<cdiv((size_t)N * HID, 256), 256, 0, stream>>>(AG, 0.0f, N * HID);
    k_attn_agg<<<cdiv((size_t)E * HID, 256), 256, 0, stream>>>(src, dst, L, DEN, V, AG, E);

    // ---- FiLM + pooling (fused) ----
    k_film<<<cdiv((size_t)B * HID, 256), 256, 0, stream>>>(wobj, W_gam, b_gam, W_bet, b_bet, GAM, BET, B);
    k_fill<<<cdiv((size_t)B * HID, 256), 256, 0, stream>>>(S, 0.0f, B * HID);
    k_fill<<<1, 256, 0, stream>>>(CNT, 0.0f, B);
    k_cnt<<<cdiv(N, 256), 256, 0, stream>>>(batch, CNT, N);
    k_film_pool<<<cdiv((size_t)N * HID, 256), 256, 0, stream>>>(AG, T, batch, GAM, BET, S, N);

    // ---- LSTM cell + LayerNorm ----
    k_gates<<<cdiv((size_t)B * 4 * LHD, 256), 256, 0, stream>>>(S, CNT, hprev, W_ih, b_ih, W_hh, b_hh, GT, B);
    k_lstm_ln<<<B, LHD, 0, stream>>>(GT, cprev, ln_g, ln_b, HN, out_h, out_c);

    // ---- dueling heads ----
    k_head_mm<<<cdiv((size_t)B * LHD, 256), 256, 0, stream>>>(HN, W_v1, b_v1, HV1, B);
    k_head_mm<<<cdiv((size_t)B * LHD, 256), 256, 0, stream>>>(HN, W_a1, b_a1, HA1, B);
    k_final<<<cdiv(B, 64), 64, 0, stream>>>(HV1, HA1, W_v2, b_v2, W_a2, b_a2, out_q, B);
}
